// MinimalNetwork_58093727645886
// MI455X (gfx1250) — compile-verified, matrix operand order checked by reading, not run on hardware
//
#include <hip/hip_runtime.h>
#include <hip/hip_bf16.h>
#include <math.h>

// ---------------------------------------------------------------------------
// Fused e3nn-style message passing for MI455X (gfx1250, wave32, WMMA).
//  kernel 1: init_c3       -- recompute real Wigner-3j tables into d_ws
//  kernel 2: pack_weights  -- W0..W3 -> bf16, scaled/padded, WMMA-B-fragment
//                             order in d_ws (L2-resident, loaded as b128)
//  kernel 3: zero_out      -- clear output accumulator
//  kernel 4: fused_kernel  -- per 16-edge tile: radial MLP via dual-tile
//                             v_wmma_f32_16x16x32_bf16 (2 accumulators ->
//                             no hazard NOPs), R tile in LDS, cooperative
//                             K = C3*Y, fully-unrolled message on VALU,
//                             scatter via global_atomic_add_f32
// ---------------------------------------------------------------------------

typedef __bf16 v16bf __attribute__((ext_vector_type(16)));
typedef float  v8f   __attribute__((ext_vector_type(8)));

#define TILE      16
#define NTHREADS  128     // 4 waves of 32
#define FEAT      72
#define RTOT      1216
#define HID       100
#define NBASIS    10
#define YLD       28      // padded sY row stride (floats)
#define SWISH_SC  1.679177f

// packed-weight fragment offsets (halfword units; 512 halfwords per fragment)
#define WP0 0
#define WP1 (WP0 + 1 * 7 * 512)     // layer0: 1 kstep  x 7 ntiles
#define WP2 (WP1 + 4 * 7 * 512)     // layer1: 4 ksteps x 7 ntiles
#define WP3 (WP2 + 4 * 7 * 512)     // layer2: 4 ksteps x 7 ntiles
#define WPTOT (WP3 + 4 * 76 * 512)  // layer3: 4 ksteps x 76 ntiles
#define WS_WPACK_BYTE_OFF 8192      // C3 table occupies first 4900 B of d_ws

// ----- compile-time tables for the unrolled message phase ------------------
constexpr int k_c3_off[19] = {0,1,10,35,44,53,80,125,170,245,350,375,420,495,600,625,700,825,1000};
constexpr int k_pr_off[9]  = {0,64,128,192,256,448,640,704,896};
constexpr int k_pr_tst[9]  = {0,1,2,3,4,7,10,11,14};
constexpr int k_foffs[3]   = {0,8,32};

// ----- runtime-indexed tables for init_c3 ----------------------------------
__device__ __constant__ int d_c3_lo[19]  = {0,0,0,1,1,1,1,1,1,1,2,2,2,2,2,2,2,2,2};
__device__ __constant__ int d_c3_li[19]  = {0,1,2,0,1,1,1,2,2,2,0,1,1,1,2,2,2,2,2};
__device__ __constant__ int d_c3_lf[19]  = {0,1,2,1,0,1,2,1,2,3,2,1,2,3,0,1,2,3,4};
__device__ __constant__ int d_c3_off[19] = {0,1,10,35,44,53,80,125,170,245,350,375,420,495,600,625,700,825,1000};

// ===========================================================================
// Kernel 1: Wigner-3j real coupling tables (1225 floats) into workspace.
// ===========================================================================
__device__ inline double dfact(int n) {
    double r = 1.0;
    for (int i = 2; i <= n; ++i) r *= (double)i;
    return r;
}

__device__ double w3j(int j1, int j2, int j3, int m1, int m2, int m3) {
    if (m1 + m2 + m3 != 0) return 0.0;
    if (j3 < abs(j1 - j2) || j3 > j1 + j2) return 0.0;
    int t1 = j2 - m1 - j3, t2 = j1 + m2 - j3;
    int kmin = max(0, max(t1, t2));
    int kmax = min(j1 + j2 - j3, min(j1 - m1, j2 + m2));
    double s = 0.0;
    for (int k = kmin; k <= kmax; ++k) {
        double d = dfact(k) * dfact(k - t1) * dfact(k - t2) *
                   dfact(j1 + j2 - j3 - k) * dfact(j1 - m1 - k) * dfact(j2 + m2 - k);
        s += ((k & 1) ? -1.0 : 1.0) / d;
    }
    double pref = (((j1 - j2 - m3) & 1) ? -1.0 : 1.0) *
        sqrt(dfact(j1 + j2 - j3) * dfact(j1 - j2 + j3) * dfact(-j1 + j2 + j3) /
             dfact(j1 + j2 + j3 + 1) *
             dfact(j1 + m1) * dfact(j1 - m1) * dfact(j2 + m2) * dfact(j2 - m2) *
             dfact(j3 + m3) * dfact(j3 - m3));
    return pref * s;
}

__device__ void cob(int l, double* re, double* im, int ld) {
    re[l * ld + l] = 1.0;
    double s = 0.7071067811865476;
    for (int m = 1; m <= l; ++m) {
        re[(l + m) * ld + (l + m)] = (m & 1) ? -s : s;
        re[(l + m) * ld + (l - m)] = s;
        im[(l - m) * ld + (l - m)] = s;
        im[(l - m) * ld + (l + m)] = (m & 1) ? s : -s;   // -i*(-1)^m*s
    }
}

__global__ void init_c3(float* __restrict__ C3tab) {
    int t = threadIdx.x;
    if (t >= 19) return;
    int l1 = d_c3_lo[t], l2 = d_c3_li[t], l3 = d_c3_lf[t];
    int A = 2 * l1 + 1, B = 2 * l2 + 1, C = 2 * l3 + 1;

    double w[5][5][9] = {};
    for (int m1 = -l1; m1 <= l1; ++m1)
        for (int m2 = -l2; m2 <= l2; ++m2) {
            int m3 = -m1 - m2;
            if (m3 >= -l3 && m3 <= l3)
                w[m1 + l1][m2 + l2][m3 + l3] = w3j(l1, l2, l3, m1, m2, m3);
        }

    double U1re[5][5] = {}, U1im[5][5] = {};
    double U2re[5][5] = {}, U2im[5][5] = {};
    double U3re[9][9] = {}, U3im[9][9] = {};
    cob(l1, &U1re[0][0], &U1im[0][0], 5);
    cob(l2, &U2re[0][0], &U2im[0][0], 5);
    cob(l3, &U3re[0][0], &U3im[0][0], 9);

    double Tre[5][5][9], Tim[5][5][9];
    double nr = 0.0, ni = 0.0;
    for (int a = 0; a < A; ++a)
        for (int b = 0; b < B; ++b)
            for (int c = 0; c < C; ++c) {
                double tre = 0.0, tim = 0.0;
                for (int m = 0; m < A; ++m)
                    for (int n = 0; n < B; ++n)
                        for (int p = 0; p < C; ++p) {
                            double wv = w[m][n][p];
                            if (wv == 0.0) continue;
                            double ar = U1re[a][m], ai = U1im[a][m];
                            double br = U2re[b][n], bi = U2im[b][n];
                            double cr = U3re[c][p], ci = U3im[c][p];
                            double xr = ar * br - ai * bi;
                            double xi = ar * bi + ai * br;
                            double zr = xr * cr - xi * ci;
                            double zi = xr * ci + xi * cr;
                            tre += zr * wv;
                            tim += zi * wv;
                        }
                Tre[a][b][c] = tre;
                Tim[a][b][c] = tim;
                nr += tre * tre;
                ni += tim * tim;
            }
    bool useRe = (nr >= ni);
    double nn = sqrt(useRe ? nr : ni);
    double inv = (nn > 0.0) ? 1.0 / nn : 0.0;
    int off = d_c3_off[t];
    for (int a = 0; a < A; ++a)
        for (int b = 0; b < B; ++b)
            for (int c = 0; c < C; ++c)
                C3tab[off + (a * B + b) * C + c] =
                    (float)((useRe ? Tre[a][b][c] : Tim[a][b][c]) * inv);
}

// ===========================================================================
// Kernel 2: pack f32 weights into bf16 WMMA B-fragments (scaled + padded).
// ===========================================================================
__global__ void pack_weights(const float* __restrict__ W, __bf16* __restrict__ dst,
                             int Kact, int Nact, int ksteps, int ntiles,
                             float scale) {
    int i = blockIdx.x * blockDim.x + threadIdx.x;
    int total = ksteps * ntiles * 512;
    if (i >= total) return;
    int frag = i >> 9;
    int r    = i & 511;
    int lane = r >> 4;
    int h    = r & 15;
    int nt = frag / ksteps;
    int ks = frag - nt * ksteps;
    int n = nt * 16 + (lane & 15);
    int half = lane >> 4;
    int k = ks * 32 + (h & 7) + 8 * half + ((h >> 3) * 16);
    float v = (k < Kact && n < Nact) ? W[k * Nact + n] * scale : 0.0f;
    dst[i] = (__bf16)v;
}

// ===========================================================================
// Kernel 3: zero output accumulator
// ===========================================================================
__global__ void zero_out(float* __restrict__ p, int n) {
    int i = blockIdx.x * blockDim.x + threadIdx.x;
    if (i < n) p[i] = 0.0f;
}

// ===========================================================================
// Kernel 4: fused radial-MLP (WMMA) + tensor message + scatter
// ===========================================================================
__device__ inline v16bf load_a_frag(const __bf16* __restrict__ act, int ld,
                                    int kbase, int lane) {
    int m = lane & 15, half = lane >> 4;
    const __bf16* p = act + m * ld + kbase + 8 * half;
    union { uint4 q[2]; v16bf v; } u;
    u.q[0] = *(const uint4*)(p);
    u.q[1] = *(const uint4*)(p + 16);
    return u.v;
}

__device__ inline v16bf load_packed_frag(const __bf16* __restrict__ p) {
    union { uint4 q[2]; v16bf v; } u;
    u.q[0] = *(const uint4*)(p);
    u.q[1] = *(const uint4*)(p + 8);
    return u.v;
}

__device__ __forceinline__ void store_tile(__bf16* __restrict__ dst, int lddst,
                                           v8f c, int nt, bool do_swish,
                                           int lane) {
    int n = nt * 16 + (lane & 15);
    int half = lane >> 4;
#pragma unroll
    for (int r = 0; r < 8; ++r) {
        int m = r + 8 * half;
        float v = c[r];
        if (do_swish) {
            float sig = __builtin_amdgcn_rcpf(1.0f + __expf(-v));
            v = SWISH_SC * v * sig;
        }
        dst[m * lddst + n] = (__bf16)v;
    }
}

// One dense layer, two N-tiles in flight per wave: consecutive v_wmma ops
// target independent accumulators, so no RAW-hazard NOPs are needed.
__device__ void dense_layer(const __bf16* __restrict__ src,
                            __bf16* __restrict__ dst, int ldsrc, int lddst,
                            const __bf16* __restrict__ Wp, int ksteps,
                            int ntiles, bool do_swish, int wave, int lane) {
    v16bf afrag[4];
    for (int ks = 0; ks < ksteps; ++ks)
        afrag[ks] = load_a_frag(src, ldsrc, ks * 32, lane);

    int nt = wave;
    for (; nt + 4 < ntiles; nt += 8) {
        const __bf16* fp0 = Wp + ((nt * ksteps) << 9) + (lane << 4);
        const __bf16* fp1 = fp0 + (ksteps << 11);          // tile nt+4
        __builtin_prefetch(fp1 + (ksteps << 11), 0, 1);    // tile nt+8
        v8f c0 = {}, c1 = {};
        for (int ks = 0; ks < ksteps; ++ks) {
            v16bf b0 = load_packed_frag(fp0 + (ks << 9));
            v16bf b1 = load_packed_frag(fp1 + (ks << 9));
            c0 = __builtin_amdgcn_wmma_f32_16x16x32_bf16(
                     false, afrag[ks], false, b0, (short)0, c0, false, false);
            c1 = __builtin_amdgcn_wmma_f32_16x16x32_bf16(
                     false, afrag[ks], false, b1, (short)0, c1, false, false);
        }
        store_tile(dst, lddst, c0, nt,     do_swish, lane);
        store_tile(dst, lddst, c1, nt + 4, do_swish, lane);
    }
    if (nt < ntiles) {                                     // remainder tile
        const __bf16* fp = Wp + ((nt * ksteps) << 9) + (lane << 4);
        v8f c = {};
        for (int ks = 0; ks < ksteps; ++ks) {
            v16bf b = load_packed_frag(fp + (ks << 9));
            c = __builtin_amdgcn_wmma_f32_16x16x32_bf16(
                    false, afrag[ks], false, b, (short)0, c, false, false);
        }
        store_tile(dst, lddst, c, nt, do_swish, lane);
    }
}

// ---- cooperative K = C3 (x) Y: once per tile per (I,J) pair ---------------
// sK layout: sK[e*128 + p*(A*B) + o*B + b], bf16, block <= 125 per edge.
template<int I, int J>
__device__ __forceinline__ void coop_K(const float* __restrict__ sC3,
                                       const float* __restrict__ sY,
                                       __bf16* __restrict__ sK, int tid) {
    constexpr int A     = 2 * I + 1;
    constexpr int B     = 2 * J + 1;
    constexpr int mn    = (I < J) ? I : J;
    constexpr int nlf   = 2 * mn + 1;
    constexpr int lfmin = (I > J) ? (I - J) : (J - I);
    constexpr int AB    = A * B;
    constexpr int PV    = nlf * AB;
    for (int i = tid; i < TILE * PV; i += NTHREADS) {
        int e = i / PV;
        int q = i - e * PV;
        int p = q / AB;
        int r = q - p * AB;
        float s = 0.f;
#pragma unroll
        for (int pp = 0; pp < nlf; ++pp) {
            if (p == pp) {
                const int lf   = lfmin + pp;
                const int Cdim = 2 * lf + 1;
                const float* C = sC3 + k_c3_off[k_pr_tst[I * 3 + J] + pp] + r * Cdim;
                const float* Y = sY + e * YLD + lf * lf;
#pragma unroll
                for (int f = 0; f < Cdim; ++f) s += C[f] * Y[f];
            }
        }
        sK[e * 128 + q] = (__bf16)s;
    }
}

// ---- per-thread consume: acc[u][o] += sum_v R[u,v,p] * (Km . F_v) ---------
template<int I, int J, int P>
__device__ __forceinline__ void path_consume(const __bf16* __restrict__ sKe,
                                             const float* __restrict__ Fr,
                                             const __bf16* __restrict__ rh,
                                             float* __restrict__ acc) {
    constexpr int Adim = 2 * I + 1;
    constexpr int Bdim = 2 * J + 1;
    constexpr int mn   = (I < J) ? I : J;
    constexpr int nlf  = 2 * mn + 1;
    constexpr int foff = k_foffs[J];
    float Km[Adim][Bdim];
#pragma unroll
    for (int o = 0; o < Adim; ++o)
#pragma unroll
        for (int b = 0; b < Bdim; ++b)
            Km[o][b] = (float)sKe[P * Adim * Bdim + o * Bdim + b];
#pragma unroll
    for (int v = 0; v < 8; ++v) {
        float rv = (float)rh[v * nlf + P];
        const float* Fv = Fr + foff + v * Bdim;
#pragma unroll
        for (int o = 0; o < Adim; ++o) {
            float t = 0.f;
#pragma unroll
            for (int b = 0; b < Bdim; ++b) t += Km[o][b] * Fv[b];
            acc[o] += rv * t;
        }
    }
}

template<int I, int J>
__device__ __forceinline__ void pair_consume(const __bf16* __restrict__ sKe,
                                             const float* __restrict__ Fr,
                                             const __bf16* __restrict__ Re,
                                             int u, float* __restrict__ acc) {
    constexpr int mn   = (I < J) ? I : J;
    constexpr int nlf  = 2 * mn + 1;
    constexpr int roff = k_pr_off[I * 3 + J];
    union { uint4 q[5]; __bf16 h[40]; } rb;
    const __bf16* rp = Re + roff + u * (8 * nlf);
#pragma unroll
    for (int c = 0; c < nlf; ++c) rb.q[c] = *(const uint4*)(rp + c * 8);
    path_consume<I, J, 0>(sKe, Fr, rb.h, acc);
    if constexpr (nlf > 2) {
        path_consume<I, J, 1>(sKe, Fr, rb.h, acc);
        path_consume<I, J, 2>(sKe, Fr, rb.h, acc);
    }
    if constexpr (nlf > 4) {
        path_consume<I, J, 3>(sKe, Fr, rb.h, acc);
        path_consume<I, J, 4>(sKe, Fr, rb.h, acc);
    }
}

__global__ __launch_bounds__(NTHREADS)
void fused_kernel(const float* __restrict__ feats,
                  const int*   __restrict__ edge_index,   // [2,E]
                  const float* __restrict__ radii,
                  const float* __restrict__ rsh,
                  const __bf16* __restrict__ wpack,
                  const float* __restrict__ C3tab,
                  float* __restrict__ out, int E) {
    __shared__ __align__(16) __bf16 sAct0[TILE * 128];   // reused as sK later
    __shared__ __align__(16) __bf16 sAct1[TILE * 128];
    __shared__ __align__(16) __bf16 sR[TILE * RTOT];     // 38.9 KB
    __shared__ __align__(16) float  sF[TILE * FEAT];
    __shared__ __align__(16) float  sY[TILE * YLD];
    __shared__ float  sC3[1225];
    __shared__ int    sSrc[TILE];
    __shared__ int    sDst[TILE];

    int tid  = threadIdx.x;
    int lane = tid & 31;
    int wave = tid >> 5;
    int e0   = blockIdx.x * TILE;

    // ---- phase 0: zero padding, edge ids, basis, C3, F_src, Y -------------
    {
        uint4 z = {0u, 0u, 0u, 0u};
        uint4* a0 = (uint4*)sAct0;
        uint4* a1 = (uint4*)sAct1;
        for (int i = tid; i < (TILE * 128) / 8; i += NTHREADS) {
            a0[i] = z;
            a1[i] = z;
        }
    }
    if (tid < TILE) {
        int ge = e0 + tid;
        sSrc[tid] = (ge < E) ? edge_index[ge]     : 0;
        sDst[tid] = (ge < E) ? edge_index[E + ge] : 0;
    }
    for (int i = tid; i < 1225; i += NTHREADS) sC3[i] = C3tab[i];
    __syncthreads();
    for (int i = tid; i < TILE * NBASIS; i += NTHREADS) {
        int e = i / NBASIS, b = i % NBASIS;
        int ge = e0 + e;
        float v = 0.0f;
        if (ge < E) {
            float r = radii[ge];
            float c = 0.7f + 0.27777778f * (float)b;     // linspace(.7,3.2,10)
            float t = (r - c) * 3.6f;                    // 1/sigma = 9/2.5
            v = __expf(-t * t);
        }
        sAct0[e * 128 + b] = (__bf16)v;
    }
    for (int i = tid; i < TILE * FEAT; i += NTHREADS) {
        int e = i / FEAT, f = i % FEAT;
        int ge = e0 + e;
        sF[e * FEAT + f] = (ge < E) ? feats[sSrc[e] * FEAT + f] : 0.0f;
    }
    for (int i = tid; i < TILE * 25; i += NTHREADS) {
        int e = i / 25, f = i % 25;
        int ge = e0 + e;
        sY[e * YLD + f] = (ge < E) ? rsh[ge * 25 + f] : 0.0f;
    }
    __syncthreads();

    // ---- phase 1: radial MLP via WMMA -------------------------------------
    dense_layer(sAct0, sAct1, 128, 128, wpack + WP0, 1, 7,  true,  wave, lane);
    __syncthreads();
    dense_layer(sAct1, sAct0, 128, 128, wpack + WP1, 4, 7,  true,  wave, lane);
    __syncthreads();
    dense_layer(sAct0, sAct1, 128, 128, wpack + WP2, 4, 7,  true,  wave, lane);
    __syncthreads();
    dense_layer(sAct1, sR,   128, RTOT, wpack + WP3, 4, 76, false, wave, lane);

    // ---- phase 2: cooperative-K message + scatter -------------------------
    int e = tid >> 3;
    int u = tid & 7;
    int ge = e0 + e;
    bool valid = (ge < E);
    float Fr[FEAT];
    if (valid) {
        const float4* sp = (const float4*)&sF[e * FEAT];
        float4* fq = (float4*)Fr;
#pragma unroll
        for (int c = 0; c < FEAT / 4; ++c) fq[c] = sp[c];
    }
    const __bf16* Re = &sR[e * RTOT];
    __bf16* sK = sAct0;            // activations are dead; reuse as K buffer
    const __bf16* sKe = sK + e * 128;
    float a0[1] = {0.f};
    float a1[3] = {0.f, 0.f, 0.f};
    float a2[5] = {0.f, 0.f, 0.f, 0.f, 0.f};

#define DO_PAIR(II, JJ, ACC)                                       \
    __syncthreads();                                               \
    coop_K<II, JJ>(sC3, sY, sK, tid);                              \
    __syncthreads();                                               \
    if (valid) pair_consume<II, JJ>(sKe, Fr, Re, u, ACC);

    DO_PAIR(0, 0, a0)
    DO_PAIR(0, 1, a0)
    DO_PAIR(0, 2, a0)
    DO_PAIR(1, 0, a1)
    DO_PAIR(1, 1, a1)
    DO_PAIR(1, 2, a1)
    DO_PAIR(2, 0, a2)
    DO_PAIR(2, 1, a2)
    DO_PAIR(2, 2, a2)
#undef DO_PAIR

    if (valid) {
        float* ob = out + sDst[e] * FEAT;
        atomicAdd(ob + u, 0.72360125f * a0[0]);          // sqrt(4pi*1/24)
        float* op1 = ob + 8 + u * 3;
#pragma unroll
        for (int o = 0; o < 3; ++o)
            atomicAdd(op1 + o, 0.82048666f * a1[o]);     // sqrt(4pi*3/56)
        float* op2 = ob + 32 + u * 5;
#pragma unroll
        for (int o = 0; o < 5; ++o)
            atomicAdd(op2 + o, 0.93416506f * a2[o]);     // sqrt(4pi*5/72)
    }
}

// ===========================================================================
extern "C" void kernel_launch(void* const* d_in, const int* in_sizes, int n_in,
                              void* d_out, int out_size, void* d_ws,
                              size_t ws_size, hipStream_t stream) {
    const float* feats      = (const float*)d_in[0];
    const int*   edge_index = (const int*)  d_in[1];
    const float* radii      = (const float*)d_in[2];
    const float* rsh        = (const float*)d_in[3];
    const float* W0         = (const float*)d_in[4];
    const float* W1         = (const float*)d_in[5];
    const float* W2         = (const float*)d_in[6];
    const float* W3         = (const float*)d_in[7];
    float*  out   = (float*)d_out;
    float*  C3tab = (float*)d_ws;                             // 1225 floats
    __bf16* wpack = (__bf16*)((char*)d_ws + WS_WPACK_BYTE_OFF); // 376 KB

    int E = in_sizes[2];                                      // 100000

    init_c3<<<1, 32, 0, stream>>>(C3tab);
    pack_weights<<<(1 * 7  * 512 + 255) / 256, 256, 0, stream>>>(
        W0, wpack + WP0, NBASIS, HID, 1, 7, 0.31622777f);
    pack_weights<<<(4 * 7  * 512 + 255) / 256, 256, 0, stream>>>(
        W1, wpack + WP1, HID, HID, 4, 7, 0.1f);
    pack_weights<<<(4 * 7  * 512 + 255) / 256, 256, 0, stream>>>(
        W2, wpack + WP2, HID, HID, 4, 7, 0.1f);
    pack_weights<<<(4 * 76 * 512 + 255) / 256, 256, 0, stream>>>(
        W3, wpack + WP3, HID, RTOT, 4, 76, 0.1f);
    zero_out<<<(out_size + 255) / 256, 256, 0, stream>>>(out, out_size);

    int tiles = (E + TILE - 1) / TILE;                        // 6250
    fused_kernel<<<tiles, NTHREADS, 0, stream>>>(
        feats, edge_index, radii, rsh, wpack, C3tab, out, E);
}
